// PQuant_vgg9_23828478558685
// MI455X (gfx1250) — 8-bit matrix operand order repaired and checked by reading, compiled, not run on hardware
//
#include <hip/hip_runtime.h>
#include <hip/hip_bf16.h>
#include <cstdint>
#include <cstddef>


typedef int v8i __attribute__((ext_vector_type(8)));

#define BN_EPS 1e-5f
#define SA_LSB (1.0f / 15.0f)

__global__ void absmax_scale_k(const float* __restrict__ w, long n,
                               float levels, float* __restrict__ scale_out) {
  __shared__ float red[256];
  float m = 0.0f;
  for (long i = threadIdx.x; i < n; i += blockDim.x)
    m = fmaxf(m, fabsf(w[i]));
  red[threadIdx.x] = m;
  __syncthreads();
  for (int s = 128; s > 0; s >>= 1) {
    if ((int)threadIdx.x < s)
      red[threadIdx.x] = fmaxf(red[threadIdx.x], red[threadIdx.x + s]);
    __syncthreads();
  }
  if (threadIdx.x == 0) scale_out[0] = red[0] / levels;
}

__global__ void pack_wq_conv_k(const float* __restrict__ w,
                               const float* __restrict__ sp,
                               int8_t* __restrict__ wpk,
                               int O, int Cin, int G, int cg) {
  long idx = (long)blockIdx.x * blockDim.x + threadIdx.x;
  long total = (long)O * G * 192;
  if (idx >= total) return;
  int k = (int)(idx % 192);
  long t = idx / 192;
  int g = (int)(t % G);
  int o = (int)(t / G);
  int8_t q = 0;
  if (k < 144) {
    int tap = k >> 4, cl = k & 15;
    int c = g * cg + cl;
    if (cl < cg && c < Cin) {
      float qf = rintf(w[((size_t)o * Cin + c) * 9 + tap] / sp[0]);
      qf = fminf(fmaxf(qf, -7.0f), 7.0f);
      q = (int8_t)qf;
    }
  }
  wpk[idx] = q;
}

__global__ void pack_wq_lin_k(const float* __restrict__ w,
                              const float* __restrict__ sp,
                              int8_t* __restrict__ wpk,
                              int O, int K, int G) {
  long idx = (long)blockIdx.x * blockDim.x + threadIdx.x;
  long total = (long)O * G * 128;
  if (idx >= total) return;
  int k = (int)(idx & 127);
  long t = idx >> 7;
  int g = (int)(t % G);
  int o = (int)(t / G);
  int kg = g * 128 + k;
  int8_t q = 0;
  if (kg < K) {
    float qf = rintf(w[(size_t)o * K + kg] / sp[0]);
    qf = fminf(fmaxf(qf, -7.0f), 7.0f);
    q = (int8_t)qf;
  }
  wpk[idx] = q;
}

__global__ void conv1_direct_k(const float* __restrict__ x,
                               const float* __restrict__ w,
                               const float* __restrict__ sp,
                               float* __restrict__ out,
                               int B, int O, int H, int W) {
  long idx = (long)blockIdx.x * blockDim.x + threadIdx.x;
  long total = (long)B * O * H * W;
  if (idx >= total) return;
  int xw = (int)(idx % W);
  long t = idx / W;
  int yy = (int)(t % H); t /= H;
  int o  = (int)(t % O);
  int b  = (int)(t / O);
  float s = sp[0];
  float acc = 0.0f;
  #pragma unroll
  for (int c = 0; c < 3; ++c)
    #pragma unroll
    for (int ky = 0; ky < 3; ++ky)
      #pragma unroll
      for (int kx = 0; kx < 3; ++kx) {
        int y2 = yy + ky - 1, x2 = xw + kx - 1;
        if ((unsigned)y2 < (unsigned)H && (unsigned)x2 < (unsigned)W) {
          float wq = rintf(w[((o * 3 + c) * 3 + ky) * 3 + kx] / s);
          wq = fminf(fmaxf(wq, -7.0f), 7.0f) * s;
          acc += x[(((size_t)b * 3 + c) * H + y2) * W + x2] * wq;
        }
      }
  out[idx] = acc;
}

template <int CONV>
__global__ void psum_gemm_iu8_k(const uint8_t* __restrict__ act,
                                const int8_t*  __restrict__ wpk,
                                const float*   __restrict__ sp,
                                float* __restrict__ out,
                                int M, int N, int G,
                                int H, int W, int Korig) {
  const long wave = ((long)blockIdx.x * blockDim.x + threadIdx.x) >> 5;
  const int  lane = threadIdx.x & 31;
  const int  tilesN = (N + 63) >> 6;
  const long tile_m = wave / tilesN;
  const int  tile_nb = (int)(wave % tilesN);
  if (tile_m * 16 >= M) return;

  const float inv_sw = 1.0f / sp[0];
  const int half = lane >> 4;
  const int l16  = lane & 15;
  const int KCH  = CONV ? 3 : 2;
  const int GK   = CONV ? 192 : 128;
  const long Ktot = (long)G * GK;

  const int m_row = (int)(tile_m * 16) + l16;
  const uint8_t* abase;
  int tapoff[9];
  if (CONV) {
    const int Cpad = G * 16;
    const int Wp = W + 2;
    const int hw = H * W;
    int b = m_row / hw;
    int rem = m_row - b * hw;
    int y = rem / W;
    int x = rem - y * W;
    abase = act + (((size_t)b * (H + 2) + y) * Wp + x) * Cpad + 8 * half;
    #pragma unroll
    for (int ky = 0; ky < 3; ++ky)
      #pragma unroll
      for (int kx = 0; kx < 3; ++kx)
        tapoff[ky * 3 + kx] = (ky * Wp + kx) * Cpad;
  } else {
    abase = act + (size_t)m_row * Korig + 8 * half;
  }

  int  nn[4];
  bool nv[4];
  const uint8_t* brow[4];
  #pragma unroll
  for (int t = 0; t < 4; ++t) {
    nn[t] = tile_nb * 64 + t * 16 + l16;
    nv[t] = (nn[t] < N);
    brow[t] = (const uint8_t*)wpk + (size_t)(nv[t] ? nn[t] : 0) * Ktot +
              8 * half;
  }

  float fsum[4][8];
  #pragma unroll
  for (int t = 0; t < 4; ++t)
    #pragma unroll
    for (int r = 0; r < 8; ++r) fsum[t][r] = 0.0f;

  for (int g = 0; g < G; ++g) {
    const uint8_t* agroup = CONV ? (abase + g * 16)
                                 : (abase + (size_t)g * 128);
    v8i acc[4];

    #pragma unroll
    for (int hk = 0; hk < KCH; ++hk) {
      v8i afrag;
      #pragma unroll
      for (int p = 0; p < 4; ++p) {
        uint2 dv = make_uint2(0u, 0u);
        if (CONV) {
          const int tap = hk * 4 + p;
          if (tap < 9)
            dv = *(const uint2*)(agroup + tapoff[tap]);
        } else {
          dv = *(const uint2*)(agroup + hk * 64 + 16 * p);
        }
        afrag[2 * p]     = (int)dv.x;
        afrag[2 * p + 1] = (int)dv.y;
      }
      #pragma unroll
      for (int t = 0; t < 4; ++t) {
        const uint8_t* bp = brow[t] + (size_t)g * GK + hk * 64;
        uint2 b0 = *(const uint2*)(bp);
        uint2 b1 = *(const uint2*)(bp + 16);
        uint2 b2 = *(const uint2*)(bp + 32);
        uint2 b3 = *(const uint2*)(bp + 48);
        v8i bfrag;
        bfrag[0] = (int)b0.x; bfrag[1] = (int)b0.y;
        bfrag[2] = (int)b1.x; bfrag[3] = (int)b1.y;
        bfrag[4] = (int)b2.x; bfrag[5] = (int)b2.y;
        bfrag[6] = (int)b3.x; bfrag[7] = (int)b3.y;
        if (hk == 0)
          acc[t] = __builtin_amdgcn_wmma_i32_16x16x64_iu8(
                       false, afrag, true, bfrag, (v8i){}, false, false);
        else
          acc[t] = __builtin_amdgcn_wmma_i32_16x16x64_iu8(
                       false, afrag, true, bfrag, acc[t], false, false);
      }
    }
    if (g + 1 < G) {
      #pragma unroll
      for (int t = 0; t < 4; ++t)
        __builtin_prefetch((const void*)(brow[t] + (size_t)(g + 1) * GK), 0, 1);
    }
    #pragma unroll
    for (int t = 0; t < 4; ++t)
      #pragma unroll
      for (int r = 0; r < 8; ++r)
        fsum[t][r] += rintf((float)acc[t][r] * inv_sw);
  }

  const float lsb = sp[0] * SA_LSB;
  const int hw = H * W;
  #pragma unroll
  for (int t = 0; t < 4; ++t) {
    if (!nv[t]) continue;
    #pragma unroll
    for (int r = 0; r < 8; ++r) {
      long m = tile_m * 16 + r + 8 * half;
      float val = fsum[t][r] * lsb;
      if (CONV) {
        int b = (int)(m / hw);
        int rem = (int)(m - (long)b * hw);
        int y = rem / W, x2 = rem - y * W;
        out[(((size_t)b * N + nn[t]) * H + y) * W + x2] = val;
      } else {
        out[(size_t)m * N + nn[t]] = val;
      }
    }
  }
}

__global__ void bn_stats_k(const float* __restrict__ x,
                           float* __restrict__ mean, float* __restrict__ var,
                           int B, int C, int HW) {
  const int c = blockIdx.x;
  __shared__ float s1[256], s2[256];
  float a = 0.0f, q = 0.0f;
  long n = (long)B * HW;
  for (long i = threadIdx.x; i < n; i += 256) {
    long b = i / HW;
    long r = i - b * HW;
    float v = x[((size_t)b * C + c) * HW + r];
    a += v;
    q += v * v;
  }
  s1[threadIdx.x] = a;
  s2[threadIdx.x] = q;
  __syncthreads();
  for (int s = 128; s > 0; s >>= 1) {
    if ((int)threadIdx.x < s) {
      s1[threadIdx.x] += s1[threadIdx.x + s];
      s2[threadIdx.x] += s2[threadIdx.x + s];
    }
    __syncthreads();
  }
  if (threadIdx.x == 0) {
    float mu = s1[0] / (float)n;
    mean[c] = mu;
    var[c] = s2[0] / (float)n - mu * mu;
  }
}

__global__ void bn_act_quant_pad_k(const float* __restrict__ x,
                                   const float* __restrict__ gb,
                                   const float* __restrict__ mean,
                                   const float* __restrict__ var,
                                   uint8_t* __restrict__ out,
                                   int B, int C, int H, int W,
                                   int G, int cg) {
  const int Cpad = G * 16, Hp = H + 2, Wp = W + 2;
  long total = (long)B * Hp * Wp * Cpad;
  long idx = (long)blockIdx.x * blockDim.x + threadIdx.x;
  if (idx >= total) return;
  int cs = (int)(idx % Cpad);
  long t = idx / Cpad;
  int xp = (int)(t % Wp); t /= Wp;
  int yp = (int)(t % Hp);
  int b  = (int)(t / Hp);
  int cl = cs & 15, g = cs >> 4;
  int ch = g * cg + cl;
  int y = yp - 1, xx = xp - 1;
  uint8_t v = 0;
  if (cl < cg && ch < C &&
      (unsigned)y < (unsigned)H && (unsigned)xx < (unsigned)W) {
    float val = x[(((size_t)b * C + ch) * H + y) * W + xx];
    float yv = (val - mean[ch]) * rsqrtf(var[ch] + BN_EPS) * gb[ch] + gb[C + ch];
    v = (uint8_t)rintf(fminf(fmaxf(yv, 0.0f), 1.0f) * 15.0f);
  }
  out[idx] = v;
}

__global__ void bn_act_quant_flat_k(const float* __restrict__ x,
                                    const float* __restrict__ gb,
                                    const float* __restrict__ mean,
                                    const float* __restrict__ var,
                                    uint8_t* __restrict__ out,
                                    long total, int C, int HW) {
  long idx = (long)blockIdx.x * blockDim.x + threadIdx.x;
  if (idx >= total) return;
  int c = (int)((idx / HW) % C);
  float y = (x[idx] - mean[c]) * rsqrtf(var[c] + BN_EPS) * gb[c] + gb[C + c];
  out[idx] = (uint8_t)rintf(fminf(fmaxf(y, 0.0f), 1.0f) * 15.0f);
}

__global__ void bn_apply_f32_k(const float* __restrict__ x,
                               const float* __restrict__ gb,
                               const float* __restrict__ mean,
                               const float* __restrict__ var,
                               float* __restrict__ out,
                               long total, int C, int HW) {
  long idx = (long)blockIdx.x * blockDim.x + threadIdx.x;
  if (idx >= total) return;
  int c = (int)((idx / HW) % C);
  out[idx] = (x[idx] - mean[c]) * rsqrtf(var[c] + BN_EPS) * gb[c] + gb[C + c];
}

__global__ void maxpool2_k(const float* __restrict__ in, float* __restrict__ out,
                           int B, int C, int H, int W) {
  int Ho = H >> 1, Wo = W >> 1;
  long idx = (long)blockIdx.x * blockDim.x + threadIdx.x;
  long total = (long)B * C * Ho * Wo;
  if (idx >= total) return;
  int x = (int)(idx % Wo);
  long t = idx / Wo;
  int y = (int)(t % Ho); t /= Ho;
  int c = (int)(t % C);
  int b = (int)(t / C);
  const float* p = in + (((size_t)b * C + c) * H + 2 * y) * W + 2 * x;
  out[idx] = fmaxf(fmaxf(p[0], p[1]), fmaxf(p[W], p[W + 1]));
}

__global__ void add_bias_k(float* __restrict__ x, const float* __restrict__ bias,
                           long total, int O) {
  long idx = (long)blockIdx.x * blockDim.x + threadIdx.x;
  if (idx >= total) return;
  x[idx] += bias[idx % O];
}

extern "C" void kernel_launch(void* const* d_in, const int* in_sizes, int n_in,
                              void* d_out, int out_size, void* d_ws,
                              size_t ws_size, hipStream_t stream) {
  (void)in_sizes; (void)n_in; (void)out_size; (void)ws_size;

  const float* x    = (const float*)d_in[0];
  const float* w1   = (const float*)d_in[1];
  const float* wc[5] = { (const float*)d_in[2], (const float*)d_in[3],
                         (const float*)d_in[4], (const float*)d_in[5],
                         (const float*)d_in[6] };
  const float* fc1w = (const float*)d_in[7];
  const float* fc2w = (const float*)d_in[8];
  const float* fc3w = (const float*)d_in[9];
  const float* fc3b = (const float*)d_in[10];
  const float* bn[9];
  for (int i = 0; i < 9; ++i) bn[i] = (const float*)d_in[11 + i];

  char* ws = (char*)d_ws;
  float*   F0    = (float*)(ws);
  float*   F1    = (float*)(ws + (64UL  << 20));
  uint8_t* UPAD  = (uint8_t*)(ws + (80UL << 20));
  uint8_t* UFLAT = (uint8_t*)(ws + (104UL << 20));
  int8_t*  WPK   = (int8_t*)(ws + (106UL << 20));
  float*   SC    = (float*)(ws + (114UL << 20));
  float*   MEAN  = SC + 64;
  float*   VAR   = MEAN + 1024;

  const int B = 128;
  const int cg = 14;
  auto cdiv = [](long a, long b) { return (a + b - 1) / b; };

  auto run_bn_quant_pad = [&](const float* in, const float* gb, int C, int H,
                              int W) {
    int G = (C + cg - 1) / cg;
    bn_stats_k<<<C, 256, 0, stream>>>(in, MEAN, VAR, B, C, H * W);
    long total = (long)B * (H + 2) * (W + 2) * G * 16;
    bn_act_quant_pad_k<<<cdiv(total, 256), 256, 0, stream>>>(
        in, gb, MEAN, VAR, UPAD, B, C, H, W, G, cg);
  };

  auto run_bn_quant_flat = [&](const float* in, const float* gb, int C,
                               int HW) {
    bn_stats_k<<<C, 256, 0, stream>>>(in, MEAN, VAR, B, C, HW);
    long total = (long)B * C * HW;
    bn_act_quant_flat_k<<<cdiv(total, 256), 256, 0, stream>>>(
        in, gb, MEAN, VAR, UFLAT, total, C, HW);
  };

  auto run_psum_conv = [&](const float* wf, int O, int Cin, int H, int W,
                           float* sc, float* outp) {
    const int G = (Cin + cg - 1) / cg;
    absmax_scale_k<<<1, 256, 0, stream>>>(wf, (long)O * Cin * 9, 7.0f, sc);
    long tot = (long)O * G * 192;
    pack_wq_conv_k<<<cdiv(tot, 256), 256, 0, stream>>>(wf, sc, WPK, O, Cin, G,
                                                       cg);
    long M = (long)B * H * W;
    long waves = (M / 16) * cdiv(O, 64);
    psum_gemm_iu8_k<1><<<cdiv(waves * 32, 128), 128, 0, stream>>>(
        UPAD, WPK, sc, outp, (int)M, O, G, H, W, 0);
  };

  auto run_psum_linear = [&](const float* wf, int O, int K, float* sc,
                             float* outp) {
    const int G = (K + 127) / 128;
    absmax_scale_k<<<1, 256, 0, stream>>>(wf, (long)O * K, 7.0f, sc);
    long tot = (long)O * G * 128;
    pack_wq_lin_k<<<cdiv(tot, 256), 256, 0, stream>>>(wf, sc, WPK, O, K, G);
    long waves = (B / 16) * cdiv(O, 64);
    psum_gemm_iu8_k<0><<<cdiv(waves * 32, 128), 128, 0, stream>>>(
        UFLAT, WPK, sc, outp, B, O, G, 0, 0, K);
  };

  absmax_scale_k<<<1, 256, 0, stream>>>(w1, 128L * 3 * 9, 7.0f, SC + 0);
  {
    long tot = (long)B * 128 * 32 * 32;
    conv1_direct_k<<<cdiv(tot, 256), 256, 0, stream>>>(x, w1, SC + 0, F0,
                                                       B, 128, 32, 32);
  }
  run_bn_quant_pad(F0, bn[0], 128, 32, 32);

  run_psum_conv(wc[0], 128, 128, 32, 32, SC + 1, F0);
  maxpool2_k<<<cdiv((long)B * 128 * 16 * 16, 256), 256, 0, stream>>>(
      F0, F1, B, 128, 32, 32);
  run_bn_quant_pad(F1, bn[1], 128, 16, 16);

  run_psum_conv(wc[1], 256, 128, 16, 16, SC + 2, F0);
  run_bn_quant_pad(F0, bn[2], 256, 16, 16);

  run_psum_conv(wc[2], 256, 256, 16, 16, SC + 3, F0);
  maxpool2_k<<<cdiv((long)B * 256 * 8 * 8, 256), 256, 0, stream>>>(
      F0, F1, B, 256, 16, 16);
  run_bn_quant_pad(F1, bn[3], 256, 8, 8);

  run_psum_conv(wc[3], 512, 256, 8, 8, SC + 4, F0);
  run_bn_quant_pad(F0, bn[4], 512, 8, 8);

  run_psum_conv(wc[4], 512, 512, 8, 8, SC + 5, F0);
  maxpool2_k<<<cdiv((long)B * 512 * 4 * 4, 256), 256, 0, stream>>>(
      F0, F1, B, 512, 8, 8);
  run_bn_quant_flat(F1, bn[5], 512, 4 * 4);

  run_psum_linear(fc1w, 1024, 8192, SC + 6, F0);
  run_bn_quant_flat(F0, bn[6], 1024, 1);

  run_psum_linear(fc2w, 1024, 1024, SC + 7, F0);
  run_bn_quant_flat(F0, bn[7], 1024, 1);

  run_psum_linear(fc3w, 10, 1024, SC + 8, F0);
  add_bias_k<<<cdiv((long)B * 10, 256), 256, 0, stream>>>(F0, fc3b,
                                                          (long)B * 10, 10);
  bn_stats_k<<<10, 256, 0, stream>>>(F0, MEAN, VAR, B, 10, 1);
  bn_apply_f32_k<<<cdiv((long)B * 10, 256), 256, 0, stream>>>(
      F0, bn[8], MEAN, VAR, (float*)d_out, (long)B * 10, 10, 1);
}
